// UTPM_28673201668389
// MI455X (gfx1250) — matrix operand order repaired and checked by reading, not compiled, not run on hardware
//
#include <hip/hip_runtime.h>

typedef __attribute__((ext_vector_type(16))) __bf16 v16bf;
typedef __attribute__((ext_vector_type(8)))  __bf16 v8bf;
typedef __attribute__((ext_vector_type(8)))  float  v8f;

#define EPSF 1e-5f
#define PADVAL (-4294967295.0f)

constexpr int Bn = 16384;
constexpr int Nn = 50;
constexpr int En = 100;
constexpr int Tn = 8;
constexpr int Un = 100;
constexpr int Ln = 5000;
constexpr int Pn = 4950;     // C(100,2)
constexpr int KP = 4960;     // P padded to multiple of 32 (WMMA K tiles)
constexpr int NT7 = 7;       // ceil(100/16) output col tiles for GEMM1
constexpr int NROWS7 = 112;  // padded W7t rows
constexpr int LTILES = 313;  // ceil(5000/16)
constexpr int LROWS = 5008;  // padded LEt rows

// ---------------------------------------------------------------------------
// Prep: pair table (i, j, <v_i, v_j>) in triu row-major order, zero-padded.
// ---------------------------------------------------------------------------
__global__ void k_pairs(const float* __restrict__ v,
                        unsigned char* __restrict__ gpi,
                        unsigned char* __restrict__ gpj,
                        float* __restrict__ gpw) {
  int t = blockIdx.x * blockDim.x + threadIdx.x;
  if (t < En * En) {
    int i = t / En, j = t % En;
    if (i < j) {
      int p = i * (2 * En - i - 1) / 2 + (j - i - 1);
      float w = 0.f;
      #pragma unroll
      for (int c = 0; c < 4; ++c) w += v[i * 4 + c] * v[j * 4 + c];
      gpi[p] = (unsigned char)i;
      gpj[p] = (unsigned char)j;
      gpw[p] = w;
    }
  } else if (t < En * En + (KP - Pn)) {
    int p = Pn + (t - En * En);
    gpi[p] = 0; gpj[p] = 0; gpw[p] = 0.f;   // pad contributes 0 to WMMA
  }
}

// W7 [4950][100] f32  ->  W7t [112][4960] bf16 (N-major, zero padded)
__global__ void k_w7t(const float* __restrict__ W7, __bf16* __restrict__ W7t) {
  int t = blockIdx.x * blockDim.x + threadIdx.x;
  if (t >= NROWS7 * KP) return;
  int n = t / KP, k = t % KP;
  float val = (n < En && k < Pn) ? W7[(size_t)k * En + n] : 0.f;
  W7t[(size_t)n * KP + k] = (__bf16)val;
}

// label_emb [100][5000] f32 -> LEt [5008][128] bf16 (N-major, zero padded)
__global__ void k_let(const float* __restrict__ LE, __bf16* __restrict__ LEt) {
  int t = blockIdx.x * blockDim.x + threadIdx.x;
  if (t >= LROWS * 128) return;
  int n = t / 128, k = t % 128;
  float val = (n < Ln && k < Un) ? LE[(size_t)k * Ln + n] : 0.f;
  LEt[(size_t)n * 128 + k] = (__bf16)val;
}

// ---------------------------------------------------------------------------
// BN1 stats by recompute: one block per channel n; stats of relu(x@W+b)
// over (batch, T). Emits fused scale/shift for both attention branches.
// ---------------------------------------------------------------------------
__global__ void k_bn1_stats(const float* __restrict__ x,
                            const float* __restrict__ W1, const float* __restrict__ b1,
                            const float* __restrict__ W3, const float* __restrict__ b3,
                            const float* __restrict__ g, const float* __restrict__ bb,
                            float* __restrict__ A1, float* __restrict__ C1,
                            float* __restrict__ A2, float* __restrict__ C2) {
  int n = blockIdx.x, tid = threadIdx.x;
  __shared__ float w1s[En * Tn], w3s[En * Tn];
  for (int i = tid; i < En * Tn; i += 256) { w1s[i] = W1[i]; w3s[i] = W3[i]; }
  __syncthreads();

  float s1 = 0.f, q1 = 0.f, s2 = 0.f, q2 = 0.f;
  for (int b = tid; b < Bn; b += 256) {
    const float* xr = &x[((size_t)b * Nn + n) * En];
    float acc1[Tn], acc2[Tn];
    #pragma unroll
    for (int t = 0; t < Tn; ++t) { acc1[t] = b1[t]; acc2[t] = b3[t]; }
    for (int e = 0; e < En; ++e) {
      float xv = xr[e];
      #pragma unroll
      for (int t = 0; t < Tn; ++t) {
        acc1[t] += xv * w1s[e * Tn + t];
        acc2[t] += xv * w3s[e * Tn + t];
      }
    }
    #pragma unroll
    for (int t = 0; t < Tn; ++t) {
      float v1 = fmaxf(acc1[t], 0.f); s1 += v1; q1 += v1 * v1;
      float v2 = fmaxf(acc2[t], 0.f); s2 += v2; q2 += v2 * v2;
    }
  }
  __shared__ float rs[4][256];
  rs[0][tid] = s1; rs[1][tid] = q1; rs[2][tid] = s2; rs[3][tid] = q2;
  __syncthreads();
  for (int off = 128; off > 0; off >>= 1) {
    if (tid < off) {
      #pragma unroll
      for (int a = 0; a < 4; ++a) rs[a][tid] += rs[a][tid + off];
    }
    __syncthreads();
  }
  if (tid == 0) {
    float cnt = (float)Bn * Tn;
    float m1 = rs[0][0] / cnt, v1 = rs[1][0] / cnt - m1 * m1;
    float m2 = rs[2][0] / cnt, v2 = rs[3][0] / cnt - m2 * m2;
    float sc1 = g[n] * rsqrtf(v1 + EPSF);
    float sc2 = g[n] * rsqrtf(v2 + EPSF);
    A1[n] = sc1; C1[n] = bb[n] - m1 * sc1;
    A2[n] = sc2; C2[n] = bb[n] - m2 * sc2;
  }
}

// ---------------------------------------------------------------------------
// Fused per-sample: recompute h, BN affine, scores, masked softmax,
// attention pooling (both branches), concat @ W5 + b5 -> h5_pre.
// ---------------------------------------------------------------------------
__global__ void __launch_bounds__(512)
k_pool(const float* __restrict__ x,
       const float* __restrict__ W1, const float* __restrict__ b1,
       const float* __restrict__ W3, const float* __restrict__ b3,
       const float* __restrict__ q1, const float* __restrict__ q2,
       const float* __restrict__ A1, const float* __restrict__ C1,
       const float* __restrict__ A2, const float* __restrict__ C2,
       const float* __restrict__ W5, const float* __restrict__ b5,
       float* __restrict__ h5_pre) {
  __shared__ float xs[Nn * En];       // 20 KB
  __shared__ float hb[2][Nn][Tn];     // 3.2 KB
  __shared__ float sc[2][Nn];
  __shared__ float sgnS[Nn];
  __shared__ float cat[2 * En];
  int b = blockIdx.x, tid = threadIdx.x;
  for (int i = tid; i < Nn * En; i += 512) xs[i] = x[(size_t)b * Nn * En + i];
  __syncthreads();

  if (tid < Nn * Tn) {
    int n = tid / Tn, t = tid % Tn;
    float a1 = b1[t], a2 = b3[t];
    for (int e = 0; e < En; ++e) {
      float xv = xs[n * En + e];
      a1 += xv * W1[e * Tn + t];
      a2 += xv * W3[e * Tn + t];
    }
    hb[0][n][t] = A1[n] * fmaxf(a1, 0.f) + C1[n];
    hb[1][n][t] = A2[n] * fmaxf(a2, 0.f) + C2[n];
  } else if (tid < Nn * Tn + Nn) {
    int n = tid - Nn * Tn;
    float s = 0.f;
    for (int e = 0; e < En; ++e) s += fabsf(xs[n * En + e]);
    sgnS[n] = (s > 0.f) ? 1.f : 0.f;
  }
  __syncthreads();

  if (tid < Nn) {
    int n = tid;
    float s1 = 0.f, s2 = 0.f;
    #pragma unroll
    for (int t = 0; t < Tn; ++t) {
      s1 += hb[0][n][t] * q1[t];
      s2 += hb[1][n][t] * q2[t];
    }
    bool pad = (sgnS[n] == 0.f);
    sc[0][n] = pad ? PADVAL : s1;
    sc[1][n] = pad ? PADVAL : s2;
  }
  __syncthreads();

  if (tid < 2) {                       // softmax over N=50, serial per branch
    float* s = sc[tid];
    float mx = s[0];
    for (int n = 1; n < Nn; ++n) mx = fmaxf(mx, s[n]);
    float sum = 0.f;
    for (int n = 0; n < Nn; ++n) { float e = __expf(s[n] - mx); s[n] = e; sum += e; }
    float inv = 1.f / sum;
    for (int n = 0; n < Nn; ++n) s[n] *= inv;
  }
  __syncthreads();

  if (tid < 2 * En) {                  // attention pooling, both branches
    int which = tid / En, e = tid % En;
    const float* w = sc[which];
    float p = 0.f;
    for (int n = 0; n < Nn; ++n) p += w[n] * xs[n * En + e];
    cat[tid] = p;
  }
  __syncthreads();

  if (tid < En) {                      // concat @ W5 + b5
    float acc = b5[tid];
    for (int k = 0; k < 2 * En; ++k) acc += cat[k] * W5[k * En + tid];
    h5_pre[(size_t)b * En + tid] = acc;
  }
}

// ---------------------------------------------------------------------------
// Generic per-feature batch stats over [B, E] -> fused scale/shift.
// ---------------------------------------------------------------------------
__global__ void k_bnE_stats(const float* __restrict__ pre,
                            const float* __restrict__ g, const float* __restrict__ bb,
                            float* __restrict__ a, float* __restrict__ c) {
  int e = blockIdx.x, tid = threadIdx.x;
  float s = 0.f, q = 0.f;
  for (int b = tid; b < Bn; b += 256) {
    float v = pre[(size_t)b * En + e];
    s += v; q += v * v;
  }
  __shared__ float rs[2][256];
  rs[0][tid] = s; rs[1][tid] = q;
  __syncthreads();
  for (int off = 128; off > 0; off >>= 1) {
    if (tid < off) { rs[0][tid] += rs[0][tid + off]; rs[1][tid] += rs[1][tid + off]; }
    __syncthreads();
  }
  if (tid == 0) {
    float m = rs[0][0] / Bn, var = rs[1][0] / Bn - m * m;
    float scl = g[e] * rsqrtf(var + EPSF);
    a[e] = scl; c[e] = bb[e] - m * scl;
  }
}

// h5n = bn(h5_pre); h6_pre = h5n @ W6 + b6
__global__ void k_h6(const float* __restrict__ h5_pre,
                     const float* __restrict__ a5, const float* __restrict__ c5,
                     const float* __restrict__ W6, const float* __restrict__ b6,
                     float* __restrict__ h5n, float* __restrict__ h6_pre) {
  __shared__ float hs[En];
  int b = blockIdx.x, tid = threadIdx.x;
  if (tid < En) {
    float v = a5[tid] * h5_pre[(size_t)b * En + tid] + c5[tid];
    hs[tid] = v;
    h5n[(size_t)b * En + tid] = v;
  }
  __syncthreads();
  if (tid < En) {
    float acc = b6[tid];
    for (int k = 0; k < En; ++k) acc += hs[k] * W6[k * En + tid];
    h6_pre[(size_t)b * En + tid] = acc;
  }
}

// ---------------------------------------------------------------------------
// GEMM1: h7_pre = cross(h5n) @ W7 + b7, cross built on the fly per K tile.
// Block: 128 threads (4 waves), 16 batch rows; wave -> one 16-col tile.
// ---------------------------------------------------------------------------
__global__ void __launch_bounds__(128)
k_cross_gemm(const float* __restrict__ h5n,
             const unsigned char* __restrict__ gpi,
             const unsigned char* __restrict__ gpj,
             const float* __restrict__ gpw,
             const __bf16* __restrict__ W7t,
             const float* __restrict__ b7,
             float* __restrict__ h7_pre) {
  __shared__ unsigned char lpi[KP];
  __shared__ unsigned char lpj[KP];
  __shared__ float lpw[KP];
  __shared__ float h5s[16 * 104];     // 16 rows, stride-padded
  int tid = threadIdx.x;
  int b0 = blockIdx.x * 16;
  for (int p = tid; p < KP; p += 128) { lpi[p] = gpi[p]; lpj[p] = gpj[p]; lpw[p] = gpw[p]; }
  for (int i = tid; i < 16 * En; i += 128) {
    int r = i / En, e = i % En;
    h5s[r * 104 + e] = h5n[(size_t)(b0 + r) * En + e];
  }
  __syncthreads();

  int wave = tid >> 5, lane = tid & 31;
  int nt = blockIdx.y * 4 + wave;
  if (nt >= NT7) return;

  int half = lane >> 4;               // K-half select
  int m = lane & 15;                  // A row within batch tile
  int n = nt * 16 + (lane & 15);      // output column (padded W7t row)
  int abase = half * 16;
  const __bf16* brow = W7t + (size_t)n * KP;
  const float* hrow = &h5s[m * 104];

  v8f acc = {0.f, 0.f, 0.f, 0.f, 0.f, 0.f, 0.f, 0.f};
  for (int ks = 0; ks < KP / 32; ++ks) {
    int k0 = ks * 32;
    __builtin_prefetch(brow + k0 + 128, 0, 1);
    v16bf afrag;
    #pragma unroll
    for (int u = 0; u < 8; ++u) {
      int k = k0 + abase + u;
      afrag[u] = (__bf16)(hrow[lpi[k]] * hrow[lpj[k]] * lpw[k]);
    }
    #pragma unroll
    for (int u = 0; u < 8; ++u) {
      int k = k0 + abase + 8 + u;
      afrag[8 + u] = (__bf16)(hrow[lpi[k]] * hrow[lpj[k]] * lpw[k]);
    }
    v16bf bfrag = *(const v16bf*)(brow + k0 + half * 16);
    acc = __builtin_amdgcn_wmma_f32_16x16x32_bf16(false, afrag, false, bfrag,
                                                  (short)0, acc, false, false);
  }
  if (n < En) {
    float bias = b7[n];
    #pragma unroll
    for (int r = 0; r < 8; ++r) {
      int row = half * 8 + r;
      h7_pre[(size_t)(b0 + row) * En + n] = acc[r] + bias;
    }
  }
}

// bn(h6_pre), bn(h7_pre), concat @ W8 + b8, LayerNorm -> h8 bf16 [B][128]
__global__ void __launch_bounds__(256)
k_head(const float* __restrict__ h6_pre, const float* __restrict__ h7_pre,
       const float* __restrict__ a6, const float* __restrict__ c6,
       const float* __restrict__ a7, const float* __restrict__ c7,
       const float* __restrict__ W8, const float* __restrict__ b8,
       const float* __restrict__ ln_g, const float* __restrict__ ln_b,
       __bf16* __restrict__ h8bf) {
  __shared__ float cat[2 * En];
  __shared__ float y[Un];
  __shared__ float red[2];
  int b = blockIdx.x, tid = threadIdx.x;
  if (tid < En) {
    cat[tid] = a6[tid] * h6_pre[(size_t)b * En + tid] + c6[tid];
  } else if (tid < 2 * En) {
    int e = tid - En;
    cat[tid] = a7[e] * h7_pre[(size_t)b * En + e] + c7[e];
  }
  __syncthreads();
  if (tid < Un) {
    float acc = b8[tid];
    for (int k = 0; k < 2 * En; ++k) acc += cat[k] * W8[k * Un + tid];
    y[tid] = acc;
  }
  __syncthreads();
  if (tid == 0) {
    float s = 0.f;
    for (int u = 0; u < Un; ++u) s += y[u];
    float mu = s / Un;
    float q = 0.f;
    for (int u = 0; u < Un; ++u) { float d = y[u] - mu; q += d * d; }
    red[0] = mu; red[1] = rsqrtf(q / Un + EPSF);
  }
  __syncthreads();
  if (tid < 128) {
    float val = 0.f;
    if (tid < Un) val = (y[tid] - red[0]) * red[1] * ln_g[tid] + ln_b[tid];
    h8bf[(size_t)b * 128 + tid] = (__bf16)val;   // K padded with zeros
  }
}

// ---------------------------------------------------------------------------
// GEMM2: out[B,5000] = h8[B,128]bf16 @ label_emb (pre-transposed bf16).
// ---------------------------------------------------------------------------
__global__ void __launch_bounds__(128)
k_out_gemm(const __bf16* __restrict__ h8bf, const __bf16* __restrict__ LEt,
           float* __restrict__ out) {
  int tid = threadIdx.x;
  int wave = tid >> 5, lane = tid & 31;
  int nt = blockIdx.y * 4 + wave;
  if (nt >= LTILES) return;
  int b0 = blockIdx.x * 16;
  int half = lane >> 4;
  int m = lane & 15;
  int n = nt * 16 + (lane & 15);
  const __bf16* arow = h8bf + (size_t)(b0 + m) * 128;
  const __bf16* brow = LEt + (size_t)n * 128;

  v8f acc = {0.f, 0.f, 0.f, 0.f, 0.f, 0.f, 0.f, 0.f};
  #pragma unroll
  for (int ks = 0; ks < 4; ++ks) {
    int k0 = ks * 32;
    v8bf alo = *(const v8bf*)(arow + k0 + half * 16);
    v8bf ahi = *(const v8bf*)(arow + k0 + half * 16 + 8);
    v16bf afrag;
    #pragma unroll
    for (int u = 0; u < 8; ++u) { afrag[u] = alo[u]; afrag[8 + u] = ahi[u]; }
    v16bf bfrag = *(const v16bf*)(brow + k0 + half * 16);
    acc = __builtin_amdgcn_wmma_f32_16x16x32_bf16(false, afrag, false, bfrag,
                                                  (short)0, acc, false, false);
  }
  if (n < Ln) {
    #pragma unroll
    for (int r = 0; r < 8; ++r) {
      int row = half * 8 + r;
      out[(size_t)(b0 + row) * Ln + n] = acc[r];
    }
  }
}

// ---------------------------------------------------------------------------
extern "C" void kernel_launch(void* const* d_in, const int* in_sizes, int n_in,
                              void* d_out, int out_size, void* d_ws, size_t ws_size,
                              hipStream_t stream) {
  (void)in_sizes; (void)n_in; (void)out_size; (void)ws_size;
  const float* x     = (const float*)d_in[0];
  const float* W1    = (const float*)d_in[1];
  const float* b1    = (const float*)d_in[2];
  const float* q1    = (const float*)d_in[3];
  const float* W3    = (const float*)d_in[4];
  const float* b3    = (const float*)d_in[5];
  const float* q2    = (const float*)d_in[6];
  const float* W5    = (const float*)d_in[7];
  const float* b5    = (const float*)d_in[8];
  const float* W6    = (const float*)d_in[9];
  const float* b6    = (const float*)d_in[10];
  const float* v     = (const float*)d_in[11];
  const float* W7    = (const float*)d_in[12];
  const float* b7    = (const float*)d_in[13];
  const float* W8    = (const float*)d_in[14];
  const float* b8    = (const float*)d_in[15];
  const float* bn1_g = (const float*)d_in[16];
  const float* bn1_b = (const float*)d_in[17];
  const float* bnE_g = (const float*)d_in[18];
  const float* bnE_b = (const float*)d_in[19];
  const float* ln_g  = (const float*)d_in[20];
  const float* ln_b  = (const float*)d_in[21];
  const float* LE    = (const float*)d_in[22];
  float* out = (float*)d_out;

  char* ws = (char*)d_ws;
  size_t off = 0;
  auto alloc = [&](size_t bytes) -> char* {
    char* p = ws + off;
    off = (off + bytes + 255) & ~(size_t)255;
    return p;
  };
  float* h5_pre = (float*)alloc((size_t)Bn * En * 4);
  float* h5n    = (float*)alloc((size_t)Bn * En * 4);
  float* h6_pre = (float*)alloc((size_t)Bn * En * 4);
  float* h7_pre = (float*)alloc((size_t)Bn * En * 4);
  __bf16* h8bf  = (__bf16*)alloc((size_t)Bn * 128 * 2);
  __bf16* W7t   = (__bf16*)alloc((size_t)NROWS7 * KP * 2);
  __bf16* LEt   = (__bf16*)alloc((size_t)LROWS * 128 * 2);
  unsigned char* gpi = (unsigned char*)alloc(KP);
  unsigned char* gpj = (unsigned char*)alloc(KP);
  float* gpw = (float*)alloc((size_t)KP * 4);
  float* A1 = (float*)alloc(Nn * 4); float* C1 = (float*)alloc(Nn * 4);
  float* A2 = (float*)alloc(Nn * 4); float* C2 = (float*)alloc(Nn * 4);
  float* A5 = (float*)alloc(En * 4); float* C5 = (float*)alloc(En * 4);
  float* A6 = (float*)alloc(En * 4); float* C6 = (float*)alloc(En * 4);
  float* A7 = (float*)alloc(En * 4); float* C7 = (float*)alloc(En * 4);

  k_pairs<<<dim3((En * En + (KP - Pn) + 255) / 256), 256, 0, stream>>>(v, gpi, gpj, gpw);
  k_w7t<<<dim3((NROWS7 * KP + 255) / 256), 256, 0, stream>>>(W7, W7t);
  k_let<<<dim3((LROWS * 128 + 255) / 256), 256, 0, stream>>>(LE, LEt);

  k_bn1_stats<<<Nn, 256, 0, stream>>>(x, W1, b1, W3, b3, bn1_g, bn1_b, A1, C1, A2, C2);
  k_pool<<<Bn, 512, 0, stream>>>(x, W1, b1, W3, b3, q1, q2, A1, C1, A2, C2, W5, b5, h5_pre);

  k_bnE_stats<<<En, 256, 0, stream>>>(h5_pre, bnE_g, bnE_b, A5, C5);
  k_h6<<<Bn, 128, 0, stream>>>(h5_pre, A5, C5, W6, b6, h5n, h6_pre);
  k_bnE_stats<<<En, 256, 0, stream>>>(h6_pre, bnE_g, bnE_b, A6, C6);

  k_cross_gemm<<<dim3(Bn / 16, 2), 128, 0, stream>>>(h5n, gpi, gpj, gpw, W7t, b7, h7_pre);
  k_bnE_stats<<<En, 256, 0, stream>>>(h7_pre, bnE_g, bnE_b, A7, C7);

  k_head<<<Bn, 256, 0, stream>>>(h6_pre, h7_pre, A6, C6, A7, C7, W8, b8, ln_g, ln_b, h8bf);
  k_out_gemm<<<dim3(Bn / 16, (LTILES + 3) / 4), 128, 0, stream>>>(h8bf, LEt, out);
}
